// MemoryAttention_13134009991699
// MI455X (gfx1250) — matrix operand order repaired and checked by reading, not compiled, not run on hardware
//
#include <hip/hip_runtime.h>
#include <hip/hip_bf16.h>

typedef __attribute__((ext_vector_type(16))) __bf16 v16bf;
typedef __attribute__((ext_vector_type(8)))  float  v8f;

#define D_MODEL 1024
#define N_HEADS 16
#define HEAD_DIM 64
#define BB 2
#define LL 2048
#define CTXN 2048
#define PERSN 256
#define SS (CTXN + PERSN + LL)   // 4352

__device__ __forceinline__ v8f wmma_bf16(v16bf a, v16bf b, v8f c) {
    return __builtin_amdgcn_wmma_f32_16x16x32_bf16(false, a, false, b, (short)0, c, false, false);
}

union U16B { uint4 q; __bf16 h[8]; };

// ---------------------------------------------------------------------------
// Weight convert + transpose: Wt[n][k] = (bf16)W[k][n].  Grid (32,32), 256 thr.
// ---------------------------------------------------------------------------
__global__ void wt_kernel(const float* __restrict__ W, __bf16* __restrict__ Wt) {
    __shared__ float t[32][33];
    int tx = threadIdx.x & 31, ty = threadIdx.x >> 5;
    int n0 = blockIdx.x * 32, k0 = blockIdx.y * 32;
    for (int i = 0; i < 32; i += 8)
        t[ty + i][tx] = W[(size_t)(k0 + ty + i) * D_MODEL + n0 + tx];
    __syncthreads();
    for (int i = 0; i < 32; i += 8)
        Wt[(size_t)(n0 + ty + i) * D_MODEL + k0 + tx] = (__bf16)t[tx][ty + i];
}

// ---------------------------------------------------------------------------
// Generic projection GEMM: C[M,1024] = A[M,1024] @ W + bias
// AMODE: 0 = A f32 (x rows)   1 = A f32 gathered from [ctx;pers;x]   2 = A bf16
// OMODE: 0 = bf16 row-major   1 = bf16 transposed-per-batch Vt[b][n][s]   2 = f32 row-major
// Block 256 thr = 8 waves; block tile 128(M) x 64(N); wave tile 16 x 64.
// ---------------------------------------------------------------------------
template<int AMODE, int OMODE>
__global__ void gemm_kernel(const float* __restrict__ Af32,
                            const __bf16* __restrict__ Abf,
                            const float* __restrict__ xin,
                            const float* __restrict__ ctxm,
                            const float* __restrict__ persm,
                            const __bf16* __restrict__ Wt,
                            const float* __restrict__ bias,
                            void* __restrict__ outp)
{
    const int lane = threadIdx.x & 31;
    const int wid  = threadIdx.x >> 5;
    const int l15  = lane & 15;
    const int half = lane >> 4;
    const int m_lane = blockIdx.y * 128 + wid * 16 + l15;  // A-frag row for this lane
    const int nb = blockIdx.x * 64;

    const float*  arow32 = nullptr;
    const __bf16* arow16 = nullptr;
    if (AMODE == 0) {
        arow32 = Af32 + (size_t)m_lane * D_MODEL;
    } else if (AMODE == 1) {
        int b = m_lane / SS, s = m_lane % SS;
        if (s < CTXN)              arow32 = ctxm  + (size_t)s * D_MODEL;
        else if (s < CTXN + PERSN) arow32 = persm + (size_t)(s - CTXN) * D_MODEL;
        else                       arow32 = xin + ((size_t)b * LL + (s - CTXN - PERSN)) * D_MODEL;
    } else {
        arow16 = Abf + (size_t)m_lane * D_MODEL;
    }

    const __bf16* brow0 = Wt + (size_t)(nb      + l15) * D_MODEL;
    const __bf16* brow1 = Wt + (size_t)(nb + 16 + l15) * D_MODEL;
    const __bf16* brow2 = Wt + (size_t)(nb + 32 + l15) * D_MODEL;
    const __bf16* brow3 = Wt + (size_t)(nb + 48 + l15) * D_MODEL;

    v8f acc0 = {}, acc1 = {}, acc2 = {}, acc3 = {};

    for (int k0 = 0; k0 < D_MODEL; k0 += 32) {
        v16bf a;
        if (AMODE != 2) {
            const float* p0 = arow32 + k0 + half * 16;
            const float* p1 = arow32 + k0 + half * 16 + 8;
            float4 f0 = *(const float4*)p0;
            float4 f1 = *(const float4*)(p0 + 4);
            float4 f2 = *(const float4*)p1;
            float4 f3 = *(const float4*)(p1 + 4);
            a[0]=(__bf16)f0.x;  a[1]=(__bf16)f0.y;  a[2]=(__bf16)f0.z;  a[3]=(__bf16)f0.w;
            a[4]=(__bf16)f1.x;  a[5]=(__bf16)f1.y;  a[6]=(__bf16)f1.z;  a[7]=(__bf16)f1.w;
            a[8]=(__bf16)f2.x;  a[9]=(__bf16)f2.y;  a[10]=(__bf16)f2.z; a[11]=(__bf16)f2.w;
            a[12]=(__bf16)f3.x; a[13]=(__bf16)f3.y; a[14]=(__bf16)f3.z; a[15]=(__bf16)f3.w;
        } else {
            U16B u0, u1;
            u0.q = *(const uint4*)(arow16 + k0 + half * 16);
            u1.q = *(const uint4*)(arow16 + k0 + half * 16 + 8);
            #pragma unroll
            for (int e = 0; e < 8; ++e) { a[e] = u0.h[e]; a[8 + e] = u1.h[e]; }
        }
        v16bf b0 = *(const v16bf*)(brow0 + k0 + half * 16);
        v16bf b1 = *(const v16bf*)(brow1 + k0 + half * 16);
        v16bf b2 = *(const v16bf*)(brow2 + k0 + half * 16);
        v16bf b3 = *(const v16bf*)(brow3 + k0 + half * 16);
        acc0 = wmma_bf16(a, b0, acc0);
        acc1 = wmma_bf16(a, b1, acc1);
        acc2 = wmma_bf16(a, b2, acc2);
        acc3 = wmma_bf16(a, b3, acc3);
    }

    const int mrow = blockIdx.y * 128 + wid * 16 + half * 8;  // C rows mrow..mrow+7
    #pragma unroll
    for (int t = 0; t < 4; ++t) {
        const v8f& acc = (t == 0) ? acc0 : (t == 1) ? acc1 : (t == 2) ? acc2 : acc3;
        const int n = nb + 16 * t + l15;
        const float bia = bias[n];
        if (OMODE == 2) {
            float* O = (float*)outp;
            #pragma unroll
            for (int r = 0; r < 8; ++r)
                O[(size_t)(mrow + r) * D_MODEL + n] = acc[r] + bia;
        } else if (OMODE == 0) {
            __bf16* O = (__bf16*)outp;
            #pragma unroll
            for (int r = 0; r < 8; ++r)
                O[(size_t)(mrow + r) * D_MODEL + n] = (__bf16)(acc[r] + bia);
        } else {  // OMODE 1: Vt[b][n][s]
            __bf16* O = (__bf16*)outp;
            #pragma unroll
            for (int r = 0; r < 8; ++r) {
                int m = mrow + r;
                int b = m / SS, s = m % SS;
                O[((size_t)b * D_MODEL + n) * SS + s] = (__bf16)(acc[r] + bia);
            }
        }
    }
}

// ---------------------------------------------------------------------------
// Flash attention, fully transposed: S^T = K·Q^T, Out^T = V^T·P^T.
// One wave = (b, h, 16-query tile); S looped in 32-key chunks.
// Per-lane: query = lane&15; per-query stats are lane-scalar, only one
// shfl_xor(16) each for max and sum per chunk.  4096 waves -> 512 blocks.
// ---------------------------------------------------------------------------
__global__ void attn_kernel(const __bf16* __restrict__ Qb,
                            const __bf16* __restrict__ Kb,
                            const __bf16* __restrict__ Vt,
                            __bf16* __restrict__ Ob)
{
    __shared__ __bf16 plds[8][16][32];   // [wave][query][key]  (P^T tile)
    const int lane = threadIdx.x & 31;
    const int wid  = threadIdx.x >> 5;
    const int l15  = lane & 15;
    const int half = lane >> 4;

    const int w   = blockIdx.x * 8 + wid;      // 0..4095
    const int b   = w >> 11;
    const int rem = w & 2047;
    const int h   = rem >> 7;
    const int q0  = (rem & 127) << 4;

    // Q as B-operand: b[e] = Q[q0+l15][h*64 + kc*32 + half*16 + e]
    const __bf16* qrow = Qb + ((size_t)b * LL + q0 + l15) * D_MODEL + h * HEAD_DIM;
    v16bf qb0 = *(const v16bf*)(qrow + half * 16);
    v16bf qb1 = *(const v16bf*)(qrow + 32 + half * 16);

    float mq = -1e30f, sq = 0.0f;          // per-query running max / sum
    v8f o0 = {}, o1 = {}, o2 = {}, o3 = {}; // Out^T tiles: dim = 16t+8*half+r, query = l15

    const __bf16* Kbase = Kb + (size_t)b * SS * D_MODEL + h * HEAD_DIM;
    const __bf16* Vbase = Vt + ((size_t)b * D_MODEL + h * HEAD_DIM) * SS;
    const float scale = 0.125f;   // 1/sqrt(64)

    for (int s0 = 0; s0 < SS; s0 += 32) {
        // ---- S^T: K(32x64) x Q^T(64x16) -> two 16x16 C tiles (rows=keys) ----
        v8f st0 = {}, st1 = {};
        {
            const __bf16* kr0 = Kbase + (size_t)(s0      + l15) * D_MODEL;
            const __bf16* kr1 = Kbase + (size_t)(s0 + 16 + l15) * D_MODEL;
            U16B u0, u1;
            v16bf a;
            u0.q = *(const uint4*)(kr0 + half * 16);
            u1.q = *(const uint4*)(kr0 + half * 16 + 8);
            #pragma unroll
            for (int e = 0; e < 8; ++e) { a[e] = u0.h[e]; a[8+e] = u1.h[e]; }
            st0 = wmma_bf16(a, qb0, st0);
            u0.q = *(const uint4*)(kr0 + 32 + half * 16);
            u1.q = *(const uint4*)(kr0 + 40 + half * 16);
            #pragma unroll
            for (int e = 0; e < 8; ++e) { a[e] = u0.h[e]; a[8+e] = u1.h[e]; }
            st0 = wmma_bf16(a, qb1, st0);
            u0.q = *(const uint4*)(kr1 + half * 16);
            u1.q = *(const uint4*)(kr1 + half * 16 + 8);
            #pragma unroll
            for (int e = 0; e < 8; ++e) { a[e] = u0.h[e]; a[8+e] = u1.h[e]; }
            st1 = wmma_bf16(a, qb0, st1);
            u0.q = *(const uint4*)(kr1 + 32 + half * 16);
            u1.q = *(const uint4*)(kr1 + 40 + half * 16);
            #pragma unroll
            for (int e = 0; e < 8; ++e) { a[e] = u0.h[e]; a[8+e] = u1.h[e]; }
            st1 = wmma_bf16(a, qb1, st1);
        }

        // ---- per-query online softmax (16 score values per lane) ----
        float s[16];
        #pragma unroll
        for (int r = 0; r < 8; ++r) { s[r] = st0[r] * scale; s[8+r] = st1[r] * scale; }
        float mx = s[0];
        #pragma unroll
        for (int i = 1; i < 16; ++i) mx = fmaxf(mx, s[i]);
        mx = fmaxf(mx, __shfl_xor(mx, 16, 32));          // combine the two key-halves
        const float newm  = fmaxf(mq, mx);
        const float alpha = __expf(mq - newm);
        U16B pk0, pk1;
        float ps = 0.0f;
        #pragma unroll
        for (int r = 0; r < 8; ++r) {
            float p = __expf(s[r] - newm);      ps += p; pk0.h[r] = (__bf16)p;  // keys 8*half+r
            float q = __expf(s[8+r] - newm);    ps += q; pk1.h[r] = (__bf16)q;  // keys 16+8*half+r
        }
        ps += __shfl_xor(ps, 16, 32);
        sq = sq * alpha + ps;
        mq = newm;
        #pragma unroll
        for (int r = 0; r < 8; ++r) { o0[r]*=alpha; o1[r]*=alpha; o2[r]*=alpha; o3[r]*=alpha; }

        // ---- P^T tile into LDS: row = query, col = key (packed 16B stores) ----
        *(uint4*)&plds[wid][l15][8 * half]      = pk0.q;
        *(uint4*)&plds[wid][l15][16 + 8 * half] = pk1.q;
        __builtin_amdgcn_wave_barrier();
        v16bf pfrag = *(const v16bf*)&plds[wid][l15][half * 16];  // B-frag: k=half*16+e, n=l15
        __builtin_amdgcn_wave_barrier();

        // ---- Out^T += V^T(64x32) x P^T(32x16): 4 dim-tiles, B-frag reused ----
        #pragma unroll
        for (int t = 0; t < 4; ++t) {
            const __bf16* vrow = Vbase + (size_t)(16 * t + l15) * SS + s0;
            U16B u0, u1;
            u0.q = *(const uint4*)(vrow + half * 16);
            u1.q = *(const uint4*)(vrow + half * 16 + 8);
            v16bf a;
            #pragma unroll
            for (int e = 0; e < 8; ++e) { a[e] = u0.h[e]; a[8+e] = u1.h[e]; }
            v8f& o = (t == 0) ? o0 : (t == 1) ? o1 : (t == 2) ? o2 : o3;
            o = wmma_bf16(a, pfrag, o);
        }
    }

    // ---- normalize (per-lane scalar) and write context: 4 packed 16B stores ----
    const float inv = 1.0f / sq;
    __bf16* orow = Ob + ((size_t)b * LL + q0 + l15) * D_MODEL + h * HEAD_DIM + half * 8;
    #pragma unroll
    for (int t = 0; t < 4; ++t) {
        const v8f& o = (t == 0) ? o0 : (t == 1) ? o1 : (t == 2) ? o2 : o3;
        U16B pk;
        #pragma unroll
        for (int r = 0; r < 8; ++r) pk.h[r] = (__bf16)(o[r] * inv);
        *(uint4*)(orow + 16 * t) = pk.q;
    }
}

// ---------------------------------------------------------------------------
extern "C" void kernel_launch(void* const* d_in, const int* in_sizes, int n_in,
                              void* d_out, int out_size, void* d_ws, size_t ws_size,
                              hipStream_t stream) {
    const float* x    = (const float*)d_in[0];
    const float* ctxm = (const float*)d_in[1];
    const float* pers = (const float*)d_in[2];
    const float* Wq   = (const float*)d_in[3];
    const float* bq   = (const float*)d_in[4];
    const float* Wk   = (const float*)d_in[5];
    const float* bk   = (const float*)d_in[6];
    const float* Wv   = (const float*)d_in[7];
    const float* bv   = (const float*)d_in[8];
    const float* Wo   = (const float*)d_in[9];
    const float* bo   = (const float*)d_in[10];

    char* p = (char*)d_ws;
    auto carve = [&](size_t bytes) {
        char* r = p;
        p += (bytes + 255) & ~(size_t)255;
        return r;
    };
    const size_t WSZ = (size_t)D_MODEL * D_MODEL * sizeof(__bf16);
    __bf16* Wqt = (__bf16*)carve(WSZ);
    __bf16* Wkt = (__bf16*)carve(WSZ);
    __bf16* Wvt = (__bf16*)carve(WSZ);
    __bf16* Wot = (__bf16*)carve(WSZ);
    __bf16* Qb  = (__bf16*)carve((size_t)BB * LL * D_MODEL * sizeof(__bf16));
    __bf16* Kb  = (__bf16*)carve((size_t)BB * SS * D_MODEL * sizeof(__bf16));
    __bf16* Vt  = (__bf16*)carve((size_t)BB * D_MODEL * SS * sizeof(__bf16));
    __bf16* Cb  = (__bf16*)carve((size_t)BB * LL * D_MODEL * sizeof(__bf16));

    dim3 blk(256);
    dim3 gT(32, 32);
    wt_kernel<<<gT, blk, 0, stream>>>(Wq, Wqt);
    wt_kernel<<<gT, blk, 0, stream>>>(Wk, Wkt);
    wt_kernel<<<gT, blk, 0, stream>>>(Wv, Wvt);
    wt_kernel<<<gT, blk, 0, stream>>>(Wo, Wot);

    // Q = x @ Wq + bq          (M = 4096, bf16 row-major)
    gemm_kernel<0, 0><<<dim3(16, (BB * LL) / 128), blk, 0, stream>>>(
        x, nullptr, nullptr, nullptr, nullptr, Wqt, bq, (void*)Qb);
    // K = ext @ Wk + bk        (M = 8704, bf16 row-major)
    gemm_kernel<1, 0><<<dim3(16, (BB * SS) / 128), blk, 0, stream>>>(
        nullptr, nullptr, x, ctxm, pers, Wkt, bk, (void*)Kb);
    // V = ext @ Wv + bv        (M = 8704, bf16 transposed Vt[b][dim][s])
    gemm_kernel<1, 1><<<dim3(16, (BB * SS) / 128), blk, 0, stream>>>(
        nullptr, nullptr, x, ctxm, pers, Wvt, bv, (void*)Vt);

    // flash attention -> context (bf16)
    attn_kernel<<<dim3((BB * N_HEADS * (LL / 16)) / 8), blk, 0, stream>>>(Qb, Kb, Vt, Cb);

    // out = context @ Wo + bo  (f32)
    gemm_kernel<2, 2><<<dim3(16, (BB * LL) / 128), blk, 0, stream>>>(
        nullptr, Cb, nullptr, nullptr, nullptr, Wot, bo, d_out);
}
